// TNO_22969485100006
// MI455X (gfx1250) — compile-verified, matrix operand order checked by reading, not run on hardware
//
#include <hip/hip_runtime.h>
#include <hip/hip_bf16.h>
#include <math.h>

// Problem constants (fixed by the reference)
#define BB 4
#define NN 2048
#define DD 512
#define HH 8
#define EE 512
#define D1 1024
#define HD 128
#define GAMMA 0.999f

typedef _Float16 v16h __attribute__((ext_vector_type(16)));
typedef float    v8f  __attribute__((ext_vector_type(8)));
typedef _Float16 h2   __attribute__((ext_vector_type(2)));

// ---------------------------------------------------------------------------
// SRMS norm (optionally followed by ReLU). One block per row.
// ---------------------------------------------------------------------------
__global__ void srms_kernel(const float* __restrict__ in, float* __restrict__ out,
                            int L, int relu) {
    __shared__ float red[256];
    const int row = blockIdx.x;
    const float* x = in + (size_t)row * L;
    float* y = out + (size_t)row * L;
    float p = 0.f;
    for (int i = threadIdx.x; i < L; i += 256) { float v = x[i]; p += v * v; }
    red[threadIdx.x] = p;
    __syncthreads();
    for (int s = 128; s > 0; s >>= 1) {
        if (threadIdx.x < s) red[threadIdx.x] += red[threadIdx.x + s];
        __syncthreads();
    }
    const float scale = 1.0f / (sqrtf(red[0] / (float)L) + 1e-8f);
    for (int i = threadIdx.x; i < L; i += 256) {
        float v = x[i] * scale;
        if (relu) v = fmaxf(v, 0.f);
        y[i] = v;
    }
}

// ---------------------------------------------------------------------------
// Generic WMMA GEMM: C[M,Nf] = epilogue(A[M,K] @ W[K,Nf] + bias)
//   mode 0: bias only
//   mode 1: bias + SiLU
//   mode 2: bias + SiLU, scatter output to (B,H,N,HD) layout
//   mode 3: bias + residual add (res has same [M,Nf] layout)
// Block tile 64x128x32, 256 threads = 8 waves, each wave -> 32x32 via
// 2x2 v_wmma_f32_16x16x32_f16. Assumes M%64==0, Nf%128==0, K%32==0.
// ---------------------------------------------------------------------------
__global__ void gemm_wmma_kernel(const float* __restrict__ A,
                                 const float* __restrict__ W,
                                 const float* __restrict__ bias,
                                 float* __restrict__ C,
                                 const float* __restrict__ res,
                                 int M, int Nf, int K, int mode) {
    __shared__ __align__(16) _Float16 As[64 * 32];    // [m][k] row-major
    __shared__ __align__(16) _Float16 Bs[128 * 32];   // [n][k] (W tile transposed)

    const int tid  = threadIdx.x;
    const int lane = tid & 31;
    const int wid  = tid >> 5;
    const int wm   = wid & 1;    // 0..1  (rows 0-31 / 32-63)
    const int wn   = wid >> 1;   // 0..3  (cols, 32 each)

    const int bm = blockIdx.x * 64;
    const int bn = blockIdx.y * 128;

    v8f acc[2][2];
    acc[0][0] = (v8f){}; acc[0][1] = (v8f){};
    acc[1][0] = (v8f){}; acc[1][1] = (v8f){};

    const int laneM = lane & 15;
    const int hiSel = (lane >> 4) & 1;   // upper half of wave

    const int nkb = K / 32;
    for (int kb = 0; kb < nkb; ++kb) {
        // ---- stage A tile: 64x32 floats -> f16, 8 per thread ----
        {
            const int r  = tid >> 2;            // 0..63
            const int c0 = (tid & 3) * 8;       // 0,8,16,24
            const float* src = A + (size_t)(bm + r) * K + kb * 32 + c0;
            if (kb + 1 < nkb) __builtin_prefetch(src + 32, 0, 0);
#pragma unroll
            for (int j = 0; j < 8; ++j)
                As[r * 32 + c0 + j] = (_Float16)src[j];
        }
        // ---- stage W tile transposed: each thread owns a K-pair (k0,k0+1)
        //      for 8 consecutive n -> packed 32-bit DS stores ----
        {
            const int k0 = (tid >> 4) * 2;      // 0,2,..,30
            const int n0 = (tid & 15) * 8;      // 0..120
            const float* src0 = W + (size_t)(kb * 32 + k0) * Nf + bn + n0;
            const float* src1 = src0 + Nf;
            if (kb + 1 < nkb) __builtin_prefetch(src0 + (size_t)32 * Nf, 0, 0);
#pragma unroll
            for (int j = 0; j < 8; ++j) {
                h2 pk;
                pk.x = (_Float16)src0[j];
                pk.y = (_Float16)src1[j];
                *(h2*)&Bs[(n0 + j) * 32 + k0] = pk;
            }
        }
        __syncthreads();

        // ---- load fragments (documented 16-bit A/B layout: K pairs 2g,2g+1
        //      contiguous per lane -> 128-bit DS reads per fragment) ----
        v16h fa[2], fb[2];
#pragma unroll
        for (int mi = 0; mi < 2; ++mi) {
            const int Mr = wm * 32 + mi * 16 + laneM;
#pragma unroll
            for (int g = 0; g < 8; ++g) {
                const int k0 = ((g >= 4) ? 16 : 0) + hiSel * 8 + 2 * (g & 3);
                h2 p = *(const h2*)&As[Mr * 32 + k0];
                fa[mi][2 * g]     = p.x;
                fa[mi][2 * g + 1] = p.y;
            }
        }
#pragma unroll
        for (int ni = 0; ni < 2; ++ni) {
            const int Nr = wn * 32 + ni * 16 + laneM;
#pragma unroll
            for (int g = 0; g < 8; ++g) {
                const int k0 = ((g >= 4) ? 16 : 0) + hiSel * 8 + 2 * (g & 3);
                h2 p = *(const h2*)&Bs[Nr * 32 + k0];
                fb[ni][2 * g]     = p.x;
                fb[ni][2 * g + 1] = p.y;
            }
        }

#pragma unroll
        for (int mi = 0; mi < 2; ++mi)
#pragma unroll
            for (int ni = 0; ni < 2; ++ni)
                acc[mi][ni] = __builtin_amdgcn_wmma_f32_16x16x32_f16(
                    false, fa[mi], false, fb[ni], (short)0, acc[mi][ni],
                    false, false);
        __syncthreads();
    }

    // ---- epilogue ----
#pragma unroll
    for (int mi = 0; mi < 2; ++mi) {
#pragma unroll
        for (int ni = 0; ni < 2; ++ni) {
#pragma unroll
            for (int r = 0; r < 8; ++r) {
                const int row = bm + wm * 32 + mi * 16 + r + hiSel * 8;
                const int col = bn + wn * 32 + ni * 16 + laneM;
                float val = acc[mi][ni][r] + bias[col];
                if (mode == 1 || mode == 2) {
                    val = val / (1.0f + __expf(-val));   // SiLU
                }
                if (mode == 3) {
                    val += res[(size_t)row * Nf + col];
                }
                if (mode == 2) {
                    // scatter (b,n, h*HD+d) -> ((b*H+h)*N + n)*HD + d
                    const int b = row / NN, n = row % NN;
                    const int h = col / HD, d = col % HD;
                    C[(((size_t)b * HH + h) * NN + n) * HD + d] = val;
                } else {
                    C[(size_t)row * Nf + col] = val;
                }
            }
        }
    }
}

// ---------------------------------------------------------------------------
// hcf[n,e] = n * Wp[e] + bp[e]
// ---------------------------------------------------------------------------
__global__ void posinit_kernel(const float* __restrict__ Wp,
                               const float* __restrict__ bp,
                               float* __restrict__ hcf) {
    int i = blockIdx.x * 256 + threadIdx.x;
    if (i >= NN * EE) return;
    int n = i / EE, e = i % EE;
    hcf[i] = (float)n * Wp[e] + bp[e];
}

// ---------------------------------------------------------------------------
// a[h,k,d] = (k==0 ? coef[0,h,d] : gamma^k * coef[k,h,d])
// coef layout: [N, H*HD]; a layout: [H, N, HD]
// ---------------------------------------------------------------------------
__global__ void build_a_kernel(const float* __restrict__ coef,
                               float* __restrict__ a) {
    int i = blockIdx.x * 256 + threadIdx.x;
    if (i >= HH * NN * HD) return;
    int h = i / (NN * HD);
    int k = (i / HD) % NN;
    int d = i % HD;
    float c = coef[(size_t)k * (HH * HD) + h * HD + d];
    float decay = (k == 0) ? 1.0f : __expf((float)k * __logf(GAMMA));
    a[i] = decay * c;
}

// ---------------------------------------------------------------------------
// Blocked causal depthwise conv along time:
//   y[b,h,t,d] = sum_{k=0..t} a[h,k,d] * v[b,h,t-k,d]
// One block = one output chunk of C_T=32 timesteps for one (b,h).
// v-tile staged via CDNA5 async memory->LDS DMA (ASYNCcnt-tracked, no VGPR
// round-trip); a-tile staged zero-padded in LDS (causal mask, branch-free).
// ---------------------------------------------------------------------------
#define C_T 32
__global__ void conv_kernel(const float* __restrict__ a,
                            const float* __restrict__ v,
                            float* __restrict__ y) {
    __shared__ __align__(16) float v_s[C_T * HD];            // 16 KB
    __shared__ __align__(16) float a_s[(2 * C_T - 1) * HD];  // 31.5 KB

    const int i = blockIdx.x;     // output chunk
    const int h = blockIdx.y;
    const int b = blockIdx.z;
    const int tid = threadIdx.x;
    const int d = tid & (HD - 1);
    const int r = tid >> 7;       // 0..1 -> t_local = r*16 + q

    const float* vb = v + (((size_t)b * HH + h) * NN) * HD;
    const float* ab = a + ((size_t)h * NN) * HD;

    float acc[16];
#pragma unroll
    for (int q = 0; q < 16; ++q) acc[q] = 0.f;

    for (int j = 0; j <= i; ++j) {
        // stage v chunk j: async global->LDS, 16B per lane, 4 issues/thread
        {
            const float* src = vb + (size_t)(j * C_T) * HD;
#pragma unroll
            for (int it = 0; it < (C_T * HD) / (4 * 256); ++it) {
                const int idx4 = it * 256 + tid;                 // b128 index
                const unsigned lds_addr =
                    (unsigned)(uintptr_t)(&v_s[idx4 * 4]);
                const unsigned long long gaddr =
                    (unsigned long long)(uintptr_t)(src + idx4 * 4);
                asm volatile("global_load_async_to_lds_b128 %0, %1, off"
                             :: "v"(lds_addr), "v"(gaddr) : "memory");
            }
        }
        // stage kernel rows k = (i-j)*C_T - (C_T-1) .. +(C_T-1), zero-padded
        const int kbase = (i - j) * C_T - (C_T - 1);
        for (int idx = tid; idx < (2 * C_T - 1) * HD; idx += 256) {
            const int kk = idx >> 7;
            const int kg = kbase + kk;
            a_s[idx] = (kg >= 0 && kg < NN) ? ab[(size_t)kg * HD + (idx & (HD - 1))]
                                            : 0.f;
        }
        asm volatile("s_wait_asynccnt 0x0" ::: "memory");
        __syncthreads();

        for (int s = 0; s < C_T; ++s) {
            const float vv = v_s[s * HD + d];
            const int base = (r * 16 - s + (C_T - 1)) * HD + d;
#pragma unroll
            for (int q = 0; q < 16; ++q)
                acc[q] = fmaf(a_s[base + q * HD], vv, acc[q]);
        }
        __syncthreads();
    }

    const int t0 = i * C_T + r * 16;
    float* yb = y + (((size_t)b * HH + h) * NN) * HD;
#pragma unroll
    for (int q = 0; q < 16; ++q)
        yb[(size_t)(t0 + q) * HD + d] = acc[q];
}

// ---------------------------------------------------------------------------
// g[b,n,h*HD+d] = u[b,n,h*HD+d] * y[b,h,n,d]
// ---------------------------------------------------------------------------
__global__ void gate_kernel(const float* __restrict__ u,
                            const float* __restrict__ y,
                            float* __restrict__ g) {
    int i = blockIdx.x * 256 + threadIdx.x;
    if (i >= BB * NN * D1) return;
    int c = i % D1;
    int bn = i / D1;
    int n = bn % NN, b = bn / NN;
    int h = c / HD, d = c % HD;
    g[i] = u[i] * y[(((size_t)b * HH + h) * NN + n) * HD + d];
}

// ---------------------------------------------------------------------------
extern "C" void kernel_launch(void* const* d_in, const int* in_sizes, int n_in,
                              void* d_out, int out_size, void* d_ws, size_t ws_size,
                              hipStream_t stream) {
    const float* x  = (const float*)d_in[0];
    const float* Wu = (const float*)d_in[1];
    const float* bu = (const float*)d_in[2];
    const float* Wv = (const float*)d_in[3];
    const float* bv = (const float*)d_in[4];
    const float* Wo = (const float*)d_in[5];
    const float* bo = (const float*)d_in[6];
    const float* Wp = (const float*)d_in[7];
    const float* bp = (const float*)d_in[8];
    const float* W1 = (const float*)d_in[9];
    const float* b1 = (const float*)d_in[10];
    const float* W2 = (const float*)d_in[11];
    const float* b2 = (const float*)d_in[12];
    const float* W3 = (const float*)d_in[13];
    const float* b3 = (const float*)d_in[14];
    const float* Wz = (const float*)d_in[15];
    const float* bz = (const float*)d_in[16];
    float* out = (float*)d_out;

    float* ws = (float*)d_ws;
    const size_t SZ_XN = (size_t)BB * NN * DD;   // 4M
    const size_t SZ_U  = (size_t)BB * NN * D1;   // 8M
    const size_t SZ_H  = (size_t)NN * EE;        // 1M
    const size_t SZ_CO = (size_t)NN * D1;        // 2M

    float* xn   = ws;                    // 4M
    float* u    = xn + SZ_XN;            // 8M
    float* v    = u + SZ_U;              // 8M  (B,H,N,HD); reused for g
    float* y    = v + SZ_U;              // 8M  (B,H,N,HD)
    float* hcfA = y + SZ_U;              // 1M
    float* hcfB = hcfA + SZ_H;           // 1M
    float* bufT = hcfB + SZ_H;           // 1M
    float* coef = bufT + SZ_H;           // 2M
    float* av   = coef + SZ_CO;          // 2M  (H,N,HD)

    const int MT = BB * NN;  // 8192 token rows

    // 1) xn = srms(x)
    srms_kernel<<<MT, 256, 0, stream>>>(x, xn, DD, 0);

    // 2) u = silu(xn @ Wu + bu)
    gemm_wmma_kernel<<<dim3(MT / 64, D1 / 128), 256, 0, stream>>>(
        xn, Wu, bu, u, nullptr, MT, D1, DD, 1);

    // 3) v = silu(xn @ Wv + bv), scattered to (B,H,N,HD)
    gemm_wmma_kernel<<<dim3(MT / 64, D1 / 128), 256, 0, stream>>>(
        xn, Wv, bv, v, nullptr, MT, D1, DD, 2);

    // 4) positional MLP
    posinit_kernel<<<(NN * EE + 255) / 256, 256, 0, stream>>>(Wp, bp, hcfA);

    const float* Ws[3] = {W1, W2, W3};
    const float* bs[3] = {b1, b2, b3};
    float* bufs[2] = {hcfA, hcfB};
    int cur = 0;
    for (int l = 0; l < 3; ++l) {
        srms_kernel<<<NN, 256, 0, stream>>>(bufs[cur], bufT, EE, 1);
        gemm_wmma_kernel<<<dim3(NN / 64, EE / 128), 256, 0, stream>>>(
            bufT, Ws[l], bs[l], bufs[cur ^ 1], nullptr, NN, EE, EE, 0);
        cur ^= 1;
    }
    srms_kernel<<<NN, 256, 0, stream>>>(bufs[cur], bufT, EE, 1);
    gemm_wmma_kernel<<<dim3(NN / 64, D1 / 128), 256, 0, stream>>>(
        bufT, Wz, bz, coef, nullptr, NN, D1, EE, 0);

    // 5) build decayed kernel a[h,k,d]
    build_a_kernel<<<(HH * NN * HD + 255) / 256, 256, 0, stream>>>(coef, av);

    // 6) causal depthwise conv: y[b,h,t,d]
    conv_kernel<<<dim3(NN / C_T, HH, BB), 256, 0, stream>>>(av, v, y);

    // 7) gate: g = u * y  (g reuses v buffer)
    gate_kernel<<<((int)SZ_U + 255) / 256, 256, 0, stream>>>(u, y, v);

    // 8) out = g @ Wo + bo + x
    gemm_wmma_kernel<<<dim3(MT / 64, DD / 128), 256, 0, stream>>>(
        v, Wo, bo, out, x, MT, DD, D1, 3);
}
